// MultiCuesTracker_7705171329073
// MI455X (gfx1250) — matrix operand order repaired and checked by reading, not compiled, not run on hardware
//
#include <hip/hip_runtime.h>

// ============================================================================
// MultiCuesTracker for MI455X (gfx1250, wave32).
//
// Pipeline (all on `stream`, sequential dependencies):
//   1. k_wconv : convert 9 LSTM weight matrices f32 (256,64) -> f16 in the
//                WMMA B-matrix lane layout (lane%16 = N, 16 packed halves = K).
//   2. k_pre   : box centers, current centers, consecutive-frame IoU sequence.
//   3. k_lstm  : 3 x (2-layer LSTM, H=64) over 1280 rows.  Recurrent GEMM
//                (K=64) via v_wmma_f32_16x16x32_f16; h state ping-pongs in
//                LDS, c state in registers; layer index is a template param
//                (no branches in the recurrent loop), A fragments hoisted to
//                registers once per timestep, gate-tile loop fully unrolled.
//                Gate nonlinearities use branch-free v_exp/v_rcp fast forms
//                (no libm tanhf divergence on the sequential critical path).
//   4. k_pair  : pairwise Score_cp / Score_cv / Score_rp  (B,N,N).
//   5. k_bneck : fused bottleneck (1x1 -> 3x3 -> 1x1 + 1x1 shortcut, ReLU)
//                in one LDS-tiled pass (32x32 tiles, halo).  Two launches.
//   6. k_mlp_h / k_mlp_e / k_rbb : Siamese embeddings + sigmoid similarity.
//   7. k_acube : A[b,i,m,n] = Sr[b,i,m]*Scpv[b,i,n]*Srbb[b,m,n]; 131 MB of
//                output -> float4 nontemporal streaming stores (HBM-bound:
//                ~5.6 us floor at 23.3 TB/s).
//
// Param flattening assumption: each setup_inputs() dict entry is one flat
// buffer, leaves in jax pytree order (dict keys sorted alphabetically):
//   lstm params : W_out, b_out, [L0: W_hh, W_ih, b], [L1: W_hh, W_ih, b]
//   conv params : b1, b2, b3, bs, w1, w2, w3, ws
//   rbb params  : b, c.{W1,W2,b1,b2}, r.{W1,W2,b1,b2}, w
// ============================================================================

typedef __attribute__((ext_vector_type(16))) _Float16 v16h;
typedef __attribute__((ext_vector_type(8)))  _Float16 v8h;
typedef __attribute__((ext_vector_type(8)))  float    v8f;
typedef __attribute__((ext_vector_type(4)))  float    v4f;

#define NB 8
#define NN 160
#define NT 6
#define NM (NB * NN)   // 1280 rows through each LSTM

// Branch-free fast nonlinearities (v_exp_f32 + v_rcp_f32, no EXEC games).
static __device__ __forceinline__ float sigf(float x) {
  return __builtin_amdgcn_rcpf(1.0f + __expf(-x));
}
static __device__ __forceinline__ float tanh_fast(float x) {
  return 2.0f * __builtin_amdgcn_rcpf(1.0f + __expf(-2.0f * x)) - 1.0f;
}

static __device__ __forceinline__ float iou_fn(const float* a, const float* b) {
  float ix = fmaxf(fminf(a[2], b[2]) - fmaxf(a[0], b[0]), 0.0f);
  float iy = fmaxf(fminf(a[3], b[3]) - fmaxf(a[1], b[1]), 0.0f);
  float inter = ix * iy;
  float aa = (a[2] - a[0]) * (a[3] - a[1]);
  float ab = (b[2] - b[0]) * (b[3] - b[1]);
  return inter / (aa + ab - inter + 1e-7f);
}

// ---------------------------------------------------------------------------
// WMMA f16 16x16x32 operand loaders (CDNA5 VGPR layouts, CDNA5 ISA §7.12.2)
// A (16x32 f16): lane = M (mod 16); halves j=0..7 -> K = 32s+hi .. +7,
//                j=8..15 -> K = 32s+16+hi .. +7, hi = (lane>=16)*8.
//                Both 8-half groups are contiguous in K -> two b128 loads.
// B (32x16 f16): lane%16 = N; 16 packed halves = K, same order as A (pairs per
//                VGPR), matching the ISA's 16-bit B-matrix description.
//                Stored pre-swizzled so each lane does two b128 loads.
// ---------------------------------------------------------------------------
static __device__ __forceinline__ v16h load_A(const _Float16* base, int rowStride,
                                              int s, int lane) {
  int m  = lane & 15;
  int hi = (lane >> 4) << 3;   // 0 or 8
  const _Float16* p = base + m * rowStride + s * 32 + hi;
  v8h lo = *(const v8h*)(p);
  v8h hp = *(const v8h*)(p + 16);
  return __builtin_shufflevector(lo, hp, 0,1,2,3,4,5,6,7,8,9,10,11,12,13,14,15);
}

static __device__ __forceinline__ v16h load_B(const _Float16* g, int tile, int s, int lane) {
  const _Float16* p = g + (((tile << 1) + s) * 32 + lane) * 16;
  v8h lo = *(const v8h*)(p);
  v8h hp = *(const v8h*)(p + 8);
  return __builtin_shufflevector(lo, hp, 0,1,2,3,4,5,6,7,8,9,10,11,12,13,14,15);
}

static __device__ __forceinline__ v8f wmma_acc(v16h a, v16h b, v8f c) {
  return __builtin_amdgcn_wmma_f32_16x16x32_f16(false, a, false, b, (short)0, c,
                                                false, false);
}

static __device__ __forceinline__ v8f v8f_zero() {
  v8f z = {0.f, 0.f, 0.f, 0.f, 0.f, 0.f, 0.f, 0.f};
  return z;
}

// ---------------------------------------------------------------------------
// 1. Weight conversion: f32 (256,64) -> f16 B-layout (16 tiles x 2 ksteps).
//    dst[((t*2+s)*32+lane)*16 + j] = W[(16t+lane%16)*64 + 32s + (j&7) + 8*(lane/16) + 16*(j/8)]
//    9 matrices: for each LSTM {pc,pv,pr}: L0.W_hh, L1.W_hh, L1.W_ih.
// ---------------------------------------------------------------------------
__global__ __launch_bounds__(256) void k_wconv(const float* __restrict__ pc,
                                               const float* __restrict__ pv,
                                               const float* __restrict__ pr,
                                               _Float16* __restrict__ dst) {
  int mat = blockIdx.y;                 // 0..8
  int lid = mat / 3, which = mat % 3;
  const float* P = (lid == 0) ? pc : ((lid == 1) ? pv : pr);
  int ninp = (lid == 0) ? 2 : ((lid == 1) ? 1 : 3);
  int nout = ninp;
  int oWhh0 = nout * 64 + nout;                  // after W_out, b_out
  int ob0   = oWhh0 + 16384 + 256 * ninp;        // after L0.W_hh, L0.W_ih
  int oWhh1 = ob0 + 256;
  int oWih1 = oWhh1 + 16384;
  int off = (which == 0) ? oWhh0 : ((which == 1) ? oWhh1 : oWih1);
  const float* W = P + off;                       // (256,64) row-major
  _Float16* D = dst + mat * 16384;
  for (int idx = blockIdx.x * 256 + threadIdx.x; idx < 16384; idx += gridDim.x * 256) {
    int j = idx & 15, ln = (idx >> 4) & 31, s = (idx >> 9) & 1, t = idx >> 10;
    D[idx] = (_Float16)W[(16 * t + (ln & 15)) * 64 + 32 * s + (j & 7) + ((ln >> 4) << 3) + ((j >> 3) << 4)];
  }
}

// ---------------------------------------------------------------------------
// 2. Preprocess: centers_past (1280,6,2), cur_xy (1280,2), iou_seq (1280,5)
// ---------------------------------------------------------------------------
__global__ __launch_bounds__(256) void k_pre(const float* __restrict__ cbpast,
                                             const float* __restrict__ cbcur,
                                             float* __restrict__ centers,
                                             float* __restrict__ curxy,
                                             float* __restrict__ iouseq) {
  int n = blockIdx.x * 256 + threadIdx.x;
  if (n >= NM) return;
  const float* bp = cbpast + n * NT * 4;
  for (int t = 0; t < NT; ++t) {
    centers[(n * NT + t) * 2 + 0] = 0.5f * (bp[t * 4 + 0] + bp[t * 4 + 2]);
    centers[(n * NT + t) * 2 + 1] = 0.5f * (bp[t * 4 + 1] + bp[t * 4 + 3]);
  }
  const float* cc = cbcur + n * 4;
  curxy[n * 2 + 0] = 0.5f * (cc[0] + cc[2]);
  curxy[n * 2 + 1] = 0.5f * (cc[1] + cc[3]);
  for (int t = 0; t < NT - 1; ++t)
    iouseq[n * (NT - 1) + t] = iou_fn(bp + t * 4, bp + (t + 1) * 4);
}

// ---------------------------------------------------------------------------
// 3. LSTM layer (LAYER is compile-time -> branch-free recurrent loop).
//    Gate order i,f,g,o.  Wave = 16 rows for all T.  Gate tiles processed in
//    groups {tg, tg+4, tg+8, tg+12} = (i,f,g,o) slices for hidden index
//    k = 16*tg + lane%16, so the c/h update is lane-local; c lives in
//    registers, h ping-pongs between two LDS buffers in A-friendly layout.
//    Returns the buffer index holding the final h.
// ---------------------------------------------------------------------------
template <int LAYER, int NINP, int TT>
static __device__ __forceinline__ int lstm_layer(const float* __restrict__ bb,     // gate bias (256)
                                                 const float* __restrict__ Wih0,   // layer-0 W_ih (256,NINP)
                                                 const float* __restrict__ X,      // layer-0 input (NM,TT,NINP)
                                                 const _Float16* __restrict__ WBh, // W_hh f16 B-layout
                                                 const _Float16* __restrict__ WBx, // layer-1 W_ih f16 B-layout
                                                 _Float16* hbufw,                  // [2][16][64]
                                                 _Float16* h1seqw,                 // [16][NT][64]
                                                 int lane, int gRow0) {
  int mcol = lane & 15;
  int hi8  = (lane >> 4) << 3;
  for (int i2 = lane; i2 < 16 * 64; i2 += 32) hbufw[i2] = (_Float16)0.f;
  float creg[4][8];
#pragma unroll
  for (int a = 0; a < 4; ++a)
#pragma unroll
    for (int v = 0; v < 8; ++v) creg[a][v] = 0.f;

  int cur = 0;
  for (int t = 0; t < TT; ++t) {
    const _Float16* Acur = hbufw + cur * (16 * 64);
    _Float16* hn = hbufw + (cur ^ 1) * (16 * 64);

    // A fragments are identical for all 16 gate tiles: hoist once per step.
    v16h a0 = load_A(Acur, 64, 0, lane);
    v16h a1 = load_A(Acur, 64, 1, lane);
    v16h ax0, ax1;
    if constexpr (LAYER == 1) {
      const _Float16* Ax = h1seqw + t * 64;          // row stride NT*64
      ax0 = load_A(Ax, NT * 64, 0, lane);
      ax1 = load_A(Ax, NT * 64, 1, lane);
    }

    float xr[8][NINP];                                // layer-0 inputs per row
    if constexpr (LAYER == 0) {
#pragma unroll
      for (int v = 0; v < 8; ++v)
#pragma unroll
        for (int kk = 0; kk < NINP; ++kk)
          xr[v][kk] = X[((gRow0 + v + hi8) * TT + t) * NINP + kk];
    }

#pragma unroll
    for (int tg = 0; tg < 4; ++tg) {
      v8f acc[4];
#pragma unroll
      for (int p2 = 0; p2 < 4; ++p2) {
        int tile = tg + 4 * p2;
        v8f a = wmma_acc(a0, load_B(WBh, tile, 0, lane), v8f_zero());
        a = wmma_acc(a1, load_B(WBh, tile, 1, lane), a);
        if constexpr (LAYER == 1) {
          a = wmma_acc(ax0, load_B(WBx, tile, 0, lane), a);
          a = wmma_acc(ax1, load_B(WBx, tile, 1, lane), a);
        }
        acc[p2] = a;
      }
      int k = 16 * tg + mcol;
#pragma unroll
      for (int p2 = 0; p2 < 4; ++p2) {
        int n = 64 * p2 + k;
        float bias = bb[n];
#pragma unroll
        for (int v = 0; v < 8; ++v) {
          float addv = bias;
          if constexpr (LAYER == 0) {
#pragma unroll
            for (int kk = 0; kk < NINP; ++kk)
              addv += xr[v][kk] * Wih0[n * NINP + kk];
          }
          acc[p2][v] += addv;
        }
      }
#pragma unroll
      for (int v = 0; v < 8; ++v) {
        float iv = acc[0][v], fv = acc[1][v], gv = acc[2][v], ov = acc[3][v];
        float c = sigf(fv) * creg[tg][v] + sigf(iv) * tanh_fast(gv);
        creg[tg][v] = c;
        float h = sigf(ov) * tanh_fast(c);
        int row = v + hi8;
        hn[row * 64 + k] = (_Float16)h;
        if constexpr (LAYER == 0) h1seqw[row * NT * 64 + t * 64 + k] = (_Float16)h;
      }
    }
    cur ^= 1;
  }
  return cur;
}

template <int NINP, int NOUT, int TT>
__global__ __launch_bounds__(128) void k_lstm(const float* __restrict__ P,
                                              const _Float16* __restrict__ WB,
                                              const float* __restrict__ X,
                                              float* __restrict__ OUT) {
  constexpr int oWhh0 = NOUT * 64 + NOUT;
  constexpr int oWih0 = oWhh0 + 16384;
  constexpr int ob0   = oWih0 + 256 * NINP;
  constexpr int oWhh1 = ob0 + 256;
  constexpr int oWih1 = oWhh1 + 16384;
  constexpr int ob1   = oWih1 + 16384;

  __shared__ __align__(16) _Float16 hbuf[4][2][16][64];     // 16 KB
  __shared__ __align__(16) _Float16 h1seq[4][16][NT][64];   // 48 KB

  int tid = threadIdx.x;
  int wave = tid >> 5, lane = tid & 31;
  int gRow0 = blockIdx.x * 64 + wave * 16;
  _Float16* hbufw  = &hbuf[wave][0][0][0];
  _Float16* h1seqw = &h1seq[wave][0][0][0];

  lstm_layer<0, NINP, TT>(P + ob0, P + oWih0, X, WB, WB + 2 * 16384,
                          hbufw, h1seqw, lane, gRow0);
  int cur = lstm_layer<1, NINP, TT>(P + ob1, P + oWih0, X, WB + 16384, WB + 2 * 16384,
                                    hbufw, h1seqw, lane, gRow0);

  // Final projection: out = h_last @ W_out^T + b_out
  const _Float16* hf = hbufw + cur * (16 * 64);
  for (int idx2 = lane; idx2 < 16 * NOUT; idx2 += 32) {
    int r = idx2 / NOUT, j = idx2 - r * NOUT;
    float s = P[NOUT * 64 + j];
#pragma unroll
    for (int kk = 0; kk < 64; ++kk) s += (float)hf[r * 64 + kk] * P[j * 64 + kk];
    OUT[(gRow0 + r) * NOUT + j] = s;
  }
}

// ---------------------------------------------------------------------------
// 4. Pairwise scores -> xfc (B,2,N,N) and xfr (B,N,N)
// ---------------------------------------------------------------------------
__global__ __launch_bounds__(256) void k_pair(const float* __restrict__ predc,
                                              const float* __restrict__ curxy,
                                              const float* __restrict__ curmask,
                                              const float* __restrict__ predr,
                                              const float* __restrict__ cbpast,
                                              const float* __restrict__ cbcur,
                                              const float* __restrict__ xrcur,
                                              float* __restrict__ xfc,
                                              float* __restrict__ xfr) {
  int idx = blockIdx.x * 256 + threadIdx.x;
  if (idx >= NB * NN * NN) return;
  int b = idx / (NN * NN);
  int r = idx - b * NN * NN;
  int i = r / NN, j = r % NN;
  int ri = b * NN + i, rj = b * NN + j;
  float d0 = predc[ri * 2 + 0] - curxy[rj * 2 + 0];
  float d1 = predc[ri * 2 + 1] - curxy[rj * 2 + 1];
  float cp = 0.5f * (d0 * d0 + d1 * d1);
  float u  = iou_fn(cbpast + (ri * NT + (NT - 1)) * 4, cbcur + rj * 4);
  float cm = curmask[ri];
  float cv = (cm - u) * (cm - u);
  xfc[(b * 2 + 0) * NN * NN + i * NN + j] = cp;
  xfc[(b * 2 + 1) * NN * NN + i * NN + j] = cv;
  float s = 0.f;
#pragma unroll
  for (int k = 0; k < 3; ++k) {
    float dd = predr[ri * 3 + k] - xrcur[rj * 3 + k];
    s += dd * dd;
  }
  xfr[idx] = s * (1.0f / 3.0f);
}

// ---------------------------------------------------------------------------
// 5. Fused bottleneck: relu(1x1) -> relu(3x3 SAME) -> 1x1 (+1x1 shortcut),
//    ReLU.  One 32x32 tile per block, halo in LDS, zero intermediates to HBM.
// ---------------------------------------------------------------------------
template <int CIN>
__global__ __launch_bounds__(256) void k_bneck(const float* __restrict__ x,
                                               const float* __restrict__ prm,
                                               float* __restrict__ out) {
  __shared__ float xs[CIN][34][36];
  __shared__ float h1s[8][34][36];
  __shared__ float pa[18 + 8 * CIN + 576 + 8 + CIN];
  constexpr int NPRM = 18 + 8 * CIN + 576 + 8 + CIN;
  int tid = threadIdx.x;
  for (int i = tid; i < NPRM; i += 256) pa[i] = prm[i];
  int blk = blockIdx.x;
  int bx = blk % 5, by = (blk / 5) % 5, b = blk / 25;
  int x0 = bx * 32, y0 = by * 32;

  for (int i = tid; i < CIN * 34 * 34; i += 256) {
    int c = i / 1156, rr = (i % 1156) / 34, q = i % 34;
    int gy = y0 - 1 + rr, gx = x0 - 1 + q;
    float v = 0.f;
    if (gy >= 0 && gy < NN && gx >= 0 && gx < NN)
      v = x[((b * CIN + c) * NN + gy) * NN + gx];
    xs[c][rr][q] = v;
  }
  __syncthreads();
  const float* w1 = pa + 18;
  const float* w2 = pa + 18 + 8 * CIN;
  const float* w3 = w2 + 576;
  const float* wsc = w3 + 8;
  for (int i = tid; i < 8 * 1156; i += 256) {
    int o = i / 1156, rr = (i % 1156) / 34, q = i % 34;
    int gy = y0 - 1 + rr, gx = x0 - 1 + q;
    float v = 0.f;  // SAME padding: h1 is zero outside the image
    if (gy >= 0 && gy < NN && gx >= 0 && gx < NN) {
      v = pa[o];
#pragma unroll
      for (int c = 0; c < CIN; ++c) v += w1[o * CIN + c] * xs[c][rr][q];
      v = fmaxf(v, 0.f);
    }
    h1s[o][rr][q] = v;
  }
  __syncthreads();
  for (int pix = tid; pix < 1024; pix += 256) {
    int py = pix >> 5, px = pix & 31;
    float h2[8];
#pragma unroll
    for (int o = 0; o < 8; ++o) {
      float s = pa[8 + o];
#pragma unroll
      for (int ci = 0; ci < 8; ++ci) {
        const float* wp = w2 + (o * 8 + ci) * 9;
        const float* h0 = &h1s[ci][py + 0][px];
        const float* h1 = &h1s[ci][py + 1][px];
        const float* hq = &h1s[ci][py + 2][px];
        s += wp[0] * h0[0] + wp[1] * h0[1] + wp[2] * h0[2];
        s += wp[3] * h1[0] + wp[4] * h1[1] + wp[5] * h1[2];
        s += wp[6] * hq[0] + wp[7] * hq[1] + wp[8] * hq[2];
      }
      h2[o] = fmaxf(s, 0.f);
    }
    float a3 = pa[16];
#pragma unroll
    for (int o = 0; o < 8; ++o) a3 += w3[o] * h2[o];
    float sc = pa[17];
#pragma unroll
    for (int c = 0; c < CIN; ++c) sc += wsc[c] * xs[c][py + 1][px + 1];
    out[(b * NN + y0 + py) * NN + (x0 + px)] = fmaxf(a3 + sc, 0.f);
  }
}

// ---------------------------------------------------------------------------
// 6. Siamese branches: hidden (relu), embedding, |diff|·w sigmoid similarity.
//    rbb param offsets: b=0, c.W1=1, c.W2=129, c.b1=2177, c.b2=2241,
//    r.W1=2273, r.W2=2465, r.b1=4513, r.b2=4577, w=4609.
// ---------------------------------------------------------------------------
__global__ __launch_bounds__(256) void k_mlp_h(const float* __restrict__ xr,
                                               const float* __restrict__ cxy,
                                               const float* __restrict__ p,
                                               float* __restrict__ hout) {
  int idx = blockIdx.x * 256 + threadIdx.x;
  if (idx >= 2 * NM * 64) return;
  int br = idx / (NM * 64);
  int rem = idx - br * NM * 64;
  int row = rem >> 6, o = rem & 63;
  float s;
  if (br == 0) {  // radar branch: din=3
    const float* W = p + 2273 + o * 3;
    const float* xi = xr + row * 3;
    s = p[4513 + o] + W[0] * xi[0] + W[1] * xi[1] + W[2] * xi[2];
  } else {        // center branch: din=2
    const float* W = p + 1 + o * 2;
    const float* xi = cxy + row * 2;
    s = p[2177 + o] + W[0] * xi[0] + W[1] * xi[1];
  }
  hout[idx] = fmaxf(s, 0.f);
}

__global__ __launch_bounds__(256) void k_mlp_e(const float* __restrict__ hout,
                                               const float* __restrict__ p,
                                               float* __restrict__ er,
                                               float* __restrict__ ec) {
  int idx = blockIdx.x * 256 + threadIdx.x;
  if (idx >= 2 * NM * 32) return;
  int br = idx / (NM * 32);
  int rem = idx - br * NM * 32;
  int row = rem >> 5, o = rem & 31;
  const float* W = p + ((br == 0) ? 2465 : 129) + o * 64;
  const float* h = hout + br * NM * 64 + row * 64;
  float s = p[((br == 0) ? 4577 : 2241) + o];
#pragma unroll
  for (int kk = 0; kk < 64; ++kk) s += W[kk] * h[kk];
  float* dst = (br == 0) ? er : ec;
  dst[rem] = s;
}

__global__ __launch_bounds__(256) void k_rbb(const float* __restrict__ er,
                                             const float* __restrict__ ec,
                                             const float* __restrict__ p,
                                             float* __restrict__ rbb) {
  int idx = blockIdx.x * 256 + threadIdx.x;
  if (idx >= NB * NN * NN) return;
  int b = idx / (NN * NN);
  int r = idx - b * NN * NN;
  int m = r / NN, n = r % NN;
  const float* a = er + (b * NN + m) * 32;
  const float* c = ec + (b * NN + n) * 32;
  float s = p[0];
#pragma unroll
  for (int e = 0; e < 32; ++e) s += p[4609 + e] * fabsf(a[e] - c[e]);
  rbb[idx] = sigf(s);
}

// ---------------------------------------------------------------------------
// 7. A_cube: 131 MB streaming write; float4 lanes + nontemporal stores.
// ---------------------------------------------------------------------------
__global__ __launch_bounds__(256) void k_acube(const float* __restrict__ scr,
                                               const float* __restrict__ cpv,
                                               const float* __restrict__ rbb,
                                               float* __restrict__ out) {
  int idx = blockIdx.x * 256 + threadIdx.x;
  if (idx >= NB * NN * NN * (NN / 4)) return;
  int n4 = (idx % 40) * 4;
  int m  = (idx / 40) % NN;
  int i  = (idx / (40 * NN)) % NN;
  int b  = idx / (40 * NN * NN);
  float r = scr[(b * NN + i) * NN + m];
  v4f c = *(const v4f*)(cpv + (b * NN + i) * NN + n4);
  v4f q = *(const v4f*)(rbb + (b * NN + m) * NN + n4);
  v4f o = r * c * q;
  __builtin_nontemporal_store(o, (v4f*)(out + ((b * NN + i) * NN + m) * NN + n4));
}

// ===========================================================================
extern "C" void kernel_launch(void* const* d_in, const int* in_sizes, int n_in,
                              void* d_out, int out_size, void* d_ws, size_t ws_size,
                              hipStream_t stream) {
  const float* Xr_past = (const float*)d_in[0];   // (8,160,6,3)
  const float* Xr_cur  = (const float*)d_in[1];   // (8,160,3)
  const float* Xcb_past = (const float*)d_in[2];  // (8,160,6,4)
  const float* Xcb_cur  = (const float*)d_in[3];  // (8,160,4)
  const float* Ppc = (const float*)d_in[4];
  const float* Ppv = (const float*)d_in[5];
  const float* Ppr = (const float*)d_in[6];
  const float* Pfc = (const float*)d_in[7];
  const float* Pfr = (const float*)d_in[8];
  const float* Prb = (const float*)d_in[9];
  float* out = (float*)d_out;

  char* w = (char*)d_ws;
  size_t off = 0;
  auto take = [&](size_t bytes) {
    size_t r = off;
    off += (bytes + 255) & ~(size_t)255;
    return r;
  };
  _Float16* wf16 = (_Float16*)(w + take(9 * 16384 * sizeof(_Float16)));   // ~288 KB
  float* centers = (float*)(w + take((size_t)NM * NT * 2 * 4));
  float* curxy   = (float*)(w + take((size_t)NM * 2 * 4));
  float* iouseq  = (float*)(w + take((size_t)NM * (NT - 1) * 4));
  float* predc   = (float*)(w + take((size_t)NM * 2 * 4));
  float* curmask = (float*)(w + take((size_t)NM * 1 * 4));
  float* predr   = (float*)(w + take((size_t)NM * 3 * 4));
  float* xfc     = (float*)(w + take((size_t)NB * 2 * NN * NN * 4));
  float* xfr     = (float*)(w + take((size_t)NB * NN * NN * 4));
  float* cpv     = (float*)(w + take((size_t)NB * NN * NN * 4));
  float* scr     = (float*)(w + take((size_t)NB * NN * NN * 4));
  float* embh    = (float*)(w + take((size_t)2 * NM * 64 * 4));
  float* er      = (float*)(w + take((size_t)NM * 32 * 4));
  float* ec      = (float*)(w + take((size_t)NM * 32 * 4));
  float* rbbS    = (float*)(w + take((size_t)NB * NN * NN * 4));
  // total ~6.3 MB of workspace

  k_wconv<<<dim3(8, 9), 256, 0, stream>>>(Ppc, Ppv, Ppr, wf16);
  k_pre<<<(NM + 255) / 256, 256, 0, stream>>>(Xcb_past, Xcb_cur, centers, curxy, iouseq);

  k_lstm<2, 2, 6><<<NM / 64, 128, 0, stream>>>(Ppc, wf16 + 0 * 16384, centers, predc);
  k_lstm<1, 1, 5><<<NM / 64, 128, 0, stream>>>(Ppv, wf16 + 3 * 16384, iouseq, curmask);
  k_lstm<3, 3, 6><<<NM / 64, 128, 0, stream>>>(Ppr, wf16 + 6 * 16384, Xr_past, predr);

  k_pair<<<(NB * NN * NN) / 256, 256, 0, stream>>>(predc, curxy, curmask, predr,
                                                   Xcb_past, Xcb_cur, Xr_cur, xfc, xfr);

  k_bneck<2><<<NB * 25, 256, 0, stream>>>(xfc, Pfc, cpv);
  k_bneck<1><<<NB * 25, 256, 0, stream>>>(xfr, Pfr, scr);

  k_mlp_h<<<(2 * NM * 64) / 256, 256, 0, stream>>>(Xr_cur, curxy, Prb, embh);
  k_mlp_e<<<(2 * NM * 32) / 256, 256, 0, stream>>>(embh, Prb, er, ec);
  k_rbb<<<(NB * NN * NN) / 256, 256, 0, stream>>>(er, ec, Prb, rbbS);

  k_acube<<<(NB * NN * NN * (NN / 4)) / 256, 256, 0, stream>>>(scr, cpv, rbbS, out);
}
